// Mamba2Block_76802605187531
// MI455X (gfx1250) — matrix operand order repaired and checked by reading, not compiled, not run on hardware
//
#include <hip/hip_runtime.h>
#include <math.h>

// ---------------------------------------------------------------------------
// Mamba2 block for MI455X (gfx1250): bf16 WMMA GEMMs + chunked SSM scan.
// B=4, S=4096, DM=768, DI=1536, H=4, N=128, K=4
// GEMM: block tile 128x256, 8 waves (2x4), wave tile 64x64 (16 WMMA / 8 loads).
// ---------------------------------------------------------------------------

typedef __bf16 bf16_t;
typedef bf16_t v16bf __attribute__((ext_vector_type(16)));
typedef bf16_t v8bf  __attribute__((ext_vector_type(8)));
typedef float  v8f   __attribute__((ext_vector_type(8)));

#define MB_ROWS 16384   // B*S
#define DM_     768
#define DI_     1536
#define HN_     512     // H*N
#define S_      4096
#define NHEAD_  4
#define NSTATE_ 128
#define NCHUNK_ 64
#define CHLEN_  64

// ---------------------------------------------------------------------------
// fp32 -> bf16 cast
// ---------------------------------------------------------------------------
__global__ void f32_to_bf16_kernel(const float* __restrict__ src,
                                   bf16_t* __restrict__ dst, long long n) {
    long long i = (long long)blockIdx.x * 256 + threadIdx.x;
    if (i < n) dst[i] = (bf16_t)src[i];
}

// ---------------------------------------------------------------------------
// WMMA GEMM:  C[M x N] = A[M x Kd] * W[N x Kd]^T (+ bias)
// A, W bf16 row-major (K contiguous).  EPI: 0 = f32 store (+bias),
// 1 = bf16 store, 2 = bf16( silu(c) * yrow[m] )
// ---------------------------------------------------------------------------
template <int EPI>
__global__ __launch_bounds__(256)
void wmma_gemm_kernel(const bf16_t* __restrict__ A, const bf16_t* __restrict__ W,
                      const float* __restrict__ bias, const float* __restrict__ yrow,
                      void* __restrict__ Cout, int M, int N, int Kd) {
    const int tid   = threadIdx.x;
    const int wid   = tid >> 5;
    const int lane  = tid & 31;
    const int half  = lane >> 4;    // which 16-lane half of the wave
    const int lr    = lane & 15;
    const int waveM = wid >> 2;     // 0..1  -> 64 rows each
    const int waveN = wid & 3;      // 0..3  -> 64 cols each
    const int wRow  = blockIdx.y * 128 + waveM * 64;
    const int wCol  = blockIdx.x * 256 + waveN * 64;

    v8f acc[4][4] = {};

    for (int k0 = 0; k0 < Kd; k0 += 32) {
        // B fragments: lane lr = column, 16 consecutive K values (32B load).
        v16bf bfrag[4];
#pragma unroll
        for (int j = 0; j < 4; ++j) {
            const bf16_t* wp = W + (size_t)(wCol + j * 16 + lr) * Kd + k0 + half * 16;
            bfrag[j] = *reinterpret_cast<const v16bf*>(wp);
        }
        // A fragments: lane lr = row; K = half*16 + 0..15.
        v16bf afrag[4];
#pragma unroll
        for (int i = 0; i < 4; ++i) {
            const bf16_t* ap = A + (size_t)(wRow + i * 16 + lr) * Kd + k0;
            v16bf a = *reinterpret_cast<const v16bf*>(ap + half * 16);
            afrag[i] = a;
        }
#pragma unroll
        for (int i = 0; i < 4; ++i)
#pragma unroll
            for (int j = 0; j < 4; ++j)
                acc[i][j] = __builtin_amdgcn_wmma_f32_16x16x32_bf16(
                    false, afrag[i], false, bfrag[j], (short)0, acc[i][j],
                    false, false);
    }

    // Epilogue. C/D layout: VGPR v -> row = half*8 + v, col = lane&15.
#pragma unroll
    for (int i = 0; i < 4; ++i) {
#pragma unroll
        for (int j = 0; j < 4; ++j) {
            const int col = wCol + j * 16 + lr;
            const float bv = bias ? bias[col] : 0.0f;
#pragma unroll
            for (int v = 0; v < 8; ++v) {
                const int row = wRow + i * 16 + half * 8 + v;
                float c = acc[i][j][v] + bv;
                const size_t o = (size_t)row * N + col;
                if (EPI == 0) {
                    ((float*)Cout)[o] = c;
                } else if (EPI == 1) {
                    ((bf16_t*)Cout)[o] = (bf16_t)c;
                } else {
                    float s = c / (1.0f + __expf(-c));   // silu
                    ((bf16_t*)Cout)[o] = (bf16_t)(s * yrow[row]);
                }
            }
        }
    }
}

// ---------------------------------------------------------------------------
// Causal depthwise conv1d (K=4), bf16 in/out, fp32 accumulate.
// ---------------------------------------------------------------------------
__global__ void conv_kernel(const bf16_t* __restrict__ U,
                            const float* __restrict__ cw,
                            const float* __restrict__ cb,
                            bf16_t* __restrict__ V) {
    long long idx = (long long)blockIdx.x * 256 + threadIdx.x;
    if (idx >= (long long)MB_ROWS * DI_) return;
    const int e = (int)(idx % DI_);
    const long long m = idx / DI_;
    const int s = (int)(m & (S_ - 1));
    float acc = cb[e];
#pragma unroll
    for (int j = 0; j < 4; ++j) {
        const int sj = s - 3 + j;
        if (sj >= 0)
            acc += (float)U[(m - 3 + j) * DI_ + e] * cw[e * 4 + j];
    }
    V[idx] = (bf16_t)acc;
}

__device__ __forceinline__ float softplus_f(float x) {
    return (x > 20.0f) ? x : log1pf(__expf(x));
}

// ---------------------------------------------------------------------------
// Scan pass A: per-chunk decay product P and local final state F (zero init).
// thread t -> n (128) | chunk (64) | h (4) | b (4);  total 131072 threads.
// ---------------------------------------------------------------------------
__global__ __launch_bounds__(256)
void scan_chunk_local(const float* __restrict__ dpre, const float* __restrict__ ssm,
                      const float* __restrict__ A_log,
                      float* __restrict__ Pw, float* __restrict__ Fw) {
    const int t  = blockIdx.x * 256 + threadIdx.x;
    const int n  = t & 127;
    const int c  = (t >> 7) & 63;
    const int hh = (t >> 13) & 3;
    const int bb = t >> 15;
    const float Av = -__expf(A_log[hh * NSTATE_ + n]);
    float P = 1.0f, F = 0.0f;
    const size_t mbase = (size_t)bb * S_ + (size_t)c * CHLEN_;
    for (int i = 0; i < CHLEN_; ++i) {
        const size_t m = mbase + i;
        const float d = softplus_f(dpre[m * HN_ + hh * NSTATE_ + n]);
        const float a = __expf(d * Av);
        const float bt = d * ssm[m * DI_ + hh * 384 + 128 + n];   // B component
        F = a * F + bt;
        P *= a;
    }
    const size_t o = (((size_t)(bb * NHEAD_ + hh) * NCHUNK_ + c) * NSTATE_) + n;
    Pw[o] = P;
    Fw[o] = F;
}

// ---------------------------------------------------------------------------
// Scan pass B: sequential carry over 64 chunks. 2048 threads.
// ---------------------------------------------------------------------------
__global__ __launch_bounds__(128)
void scan_carry(const float* __restrict__ Pw, const float* __restrict__ Fw,
                float* __restrict__ Hin) {
    const int t  = blockIdx.x * 128 + threadIdx.x;
    const int n  = t & 127;
    const int hh = (t >> 7) & 3;
    const int bb = t >> 9;
    float Hc = 0.0f;
    for (int c = 0; c < NCHUNK_; ++c) {
        const size_t o = (((size_t)(bb * NHEAD_ + hh) * NCHUNK_ + c) * NSTATE_) + n;
        Hin[o] = Hc;
        Hc = Pw[o] * Hc + Fw[o];
    }
}

// ---------------------------------------------------------------------------
// Scan pass C: replay each chunk from its carried state, produce
// y(b,s) = (1/H) sum_{h,n} C * h  via wave shuffle + LDS/global atomics.
// One block (128 threads = all n) per (b, h, chunk): 1024 blocks.
// ---------------------------------------------------------------------------
__global__ __launch_bounds__(128)
void scan_output(const float* __restrict__ dpre, const float* __restrict__ ssm,
                 const float* __restrict__ A_log, const float* __restrict__ Hin,
                 float* __restrict__ y) {
    const int g  = blockIdx.x;
    const int c  = g & 63;
    const int hh = (g >> 6) & 3;
    const int bb = g >> 8;
    const int n  = threadIdx.x;

    __shared__ float ysum[CHLEN_];
    if (n < CHLEN_) ysum[n] = 0.0f;
    __syncthreads();

    const float Av = -__expf(A_log[hh * NSTATE_ + n]);
    const size_t o = (((size_t)(bb * NHEAD_ + hh) * NCHUNK_ + c) * NSTATE_) + n;
    float h = Hin[o];
    const size_t mbase = (size_t)bb * S_ + (size_t)c * CHLEN_;

    for (int i = 0; i < CHLEN_; ++i) {
        const size_t m = mbase + i;
        const float d = softplus_f(dpre[m * HN_ + hh * NSTATE_ + n]);
        const float a = __expf(d * Av);
        const float bt = d * ssm[m * DI_ + hh * 384 + 128 + n];
        h = a * h + bt;
        float val = ssm[m * DI_ + hh * 384 + 256 + n] * h;       // C * h
#pragma unroll
        for (int off = 16; off > 0; off >>= 1)
            val += __shfl_xor(val, off, 32);
        if ((threadIdx.x & 31) == 0) atomicAdd(&ysum[i], val);
    }
    __syncthreads();
    if (n < CHLEN_) atomicAdd(&y[mbase + n], ysum[n] * 0.25f);   // sum over h / H
}

// ---------------------------------------------------------------------------
// Host launcher
// ---------------------------------------------------------------------------
extern "C" void kernel_launch(void* const* d_in, const int* in_sizes, int n_in,
                              void* d_out, int out_size, void* d_ws, size_t ws_size,
                              hipStream_t stream) {
    (void)in_sizes; (void)n_in; (void)out_size; (void)ws_size;
    const float* x      = (const float*)d_in[0];
    const float* W_in   = (const float*)d_in[1];
    const float* conv_w = (const float*)d_in[2];
    const float* conv_b = (const float*)d_in[3];
    const float* W_head = (const float*)d_in[4];
    const float* b_head = (const float*)d_in[5];
    const float* W_dt   = (const float*)d_in[6];
    const float* b_dt   = (const float*)d_in[7];
    const float* A_log  = (const float*)d_in[8];
    const float* W_gate = (const float*)d_in[9];
    const float* W_out  = (const float*)d_in[10];

    char* ws = (char*)d_ws;
    size_t off = 0;
    auto alloc = [&](size_t bytes) -> char* {
        char* p = ws + off;
        off += (bytes + 255) & ~(size_t)255;
        return p;
    };
    bf16_t* XB   = (bf16_t*)alloc((size_t)MB_ROWS * DM_ * 2);
    bf16_t* WINB = (bf16_t*)alloc((size_t)DI_ * DM_ * 2);
    bf16_t* WHB  = (bf16_t*)alloc((size_t)DI_ * DI_ * 2);
    bf16_t* WDTB = (bf16_t*)alloc((size_t)HN_ * DI_ * 2);
    bf16_t* WGB  = (bf16_t*)alloc((size_t)DI_ * DI_ * 2);
    bf16_t* WOB  = (bf16_t*)alloc((size_t)DM_ * DI_ * 2);
    bf16_t* UB   = (bf16_t*)alloc((size_t)MB_ROWS * DI_ * 2);
    bf16_t* VB   = (bf16_t*)alloc((size_t)MB_ROWS * DI_ * 2);
    bf16_t* ZB   = (bf16_t*)alloc((size_t)MB_ROWS * DI_ * 2);
    float*  SSM  = (float*)alloc((size_t)MB_ROWS * DI_ * 4);
    float*  DPRE = (float*)alloc((size_t)MB_ROWS * HN_ * 4);
    float*  PW   = (float*)alloc((size_t)4 * NHEAD_ * NCHUNK_ * NSTATE_ * 4);
    float*  FW   = (float*)alloc((size_t)4 * NHEAD_ * NCHUNK_ * NSTATE_ * 4);
    float*  HIN  = (float*)alloc((size_t)4 * NHEAD_ * NCHUNK_ * NSTATE_ * 4);
    float*  Y    = (float*)alloc((size_t)MB_ROWS * 4);

    hipMemsetAsync(Y, 0, (size_t)MB_ROWS * 4, stream);

    auto cast = [&](const float* s, bf16_t* d, long long n) {
        f32_to_bf16_kernel<<<(unsigned)((n + 255) / 256), 256, 0, stream>>>(s, d, n);
    };
    cast(x,      XB,   (long long)MB_ROWS * DM_);
    cast(W_in,   WINB, (long long)DI_ * DM_);
    cast(W_head, WHB,  (long long)DI_ * DI_);
    cast(W_dt,   WDTB, (long long)HN_ * DI_);
    cast(W_gate, WGB,  (long long)DI_ * DI_);
    cast(W_out,  WOB,  (long long)DM_ * DI_);

    // 1) U = X @ W_in^T  -> bf16
    wmma_gemm_kernel<1><<<dim3(DI_ / 256, MB_ROWS / 128), 256, 0, stream>>>(
        XB, WINB, nullptr, nullptr, UB, MB_ROWS, DI_, DM_);

    // 2) causal depthwise conv -> V (bf16)
    {
        long long n = (long long)MB_ROWS * DI_;
        conv_kernel<<<(unsigned)((n + 255) / 256), 256, 0, stream>>>(UB, conv_w, conv_b, VB);
    }

    // 3) SSM head projection (f32) and dt projection (f32)
    wmma_gemm_kernel<0><<<dim3(DI_ / 256, MB_ROWS / 128), 256, 0, stream>>>(
        VB, WHB, b_head, nullptr, SSM, MB_ROWS, DI_, DI_);
    wmma_gemm_kernel<0><<<dim3(HN_ / 256, MB_ROWS / 128), 256, 0, stream>>>(
        VB, WDTB, b_dt, nullptr, DPRE, MB_ROWS, HN_, DI_);

    // 4) chunked selective scan -> y(b,s)
    scan_chunk_local<<<512, 256, 0, stream>>>(DPRE, SSM, A_log, PW, FW);
    scan_carry<<<16, 128, 0, stream>>>(PW, FW, HIN);
    scan_output<<<1024, 128, 0, stream>>>(DPRE, SSM, A_log, HIN, Y);

    // 5) Z = silu(U @ W_gate^T) * y  -> bf16
    wmma_gemm_kernel<2><<<dim3(DI_ / 256, MB_ROWS / 128), 256, 0, stream>>>(
        UB, WGB, nullptr, Y, ZB, MB_ROWS, DI_, DI_);

    // 6) out = Z @ W_out^T -> f32 d_out
    wmma_gemm_kernel<0><<<dim3(DM_ / 256, MB_ROWS / 128), 256, 0, stream>>>(
        ZB, WOB, nullptr, nullptr, (float*)d_out, MB_ROWS, DM_, DI_);
}
